// Mamba_10187662426897
// MI455X (gfx1250) — matrix operand order repaired and checked by reading, not compiled, not run on hardware
//
#include <hip/hip_runtime.h>
#include <math.h>

// ---- model constants ----
#define NB     4
#define NTOK   256      // tokens per batch (16x16 grid)
#define DM     768
#define DI_    1536
#define NH_    24
#define HD_    64
#define DS_    128
#define CONVD  1792     // DI + 2*DS
#define DPROJ_ 3352     // 2*DI + 2*DS + NH
#define NL_    8
#define OUTC_  9

typedef __attribute__((ext_vector_type(16))) __bf16 v16bf;
typedef __attribute__((ext_vector_type(8)))  float  v8f;

union BFrag { v16bf v; unsigned short s[16]; uint4 q[2]; };

__device__ __forceinline__ unsigned short f2bf(float f) {
  union { float f; unsigned u; } c; c.f = f;
  unsigned u = c.u + 0x7FFFu + ((c.u >> 16) & 1u);   // round-to-nearest-even
  return (unsigned short)(u >> 16);
}
__device__ __forceinline__ float silu_f(float x) { return x / (1.f + __expf(-x)); }

__device__ __forceinline__ v8f wmma_bf16(const BFrag& a, const BFrag& b, v8f c) {
  return __builtin_amdgcn_wmma_f32_16x16x32_bf16(false, a.v, false, b.v, (short)0, c, false, false);
}

__device__ __forceinline__ void load_bfrag(BFrag& b, const unsigned short* p) {
  const uint4* q = (const uint4*)p;
  b.q[0] = q[0]; b.q[1] = q[1];
}

__device__ __forceinline__ void store_tile(float* __restrict__ C, const float* __restrict__ bias,
                                           v8f acc, int row0, int half8, int col, int N) {
  if (col < N) {
    const float bv = bias ? bias[col] : 0.f;
#pragma unroll
    for (int r = 0; r < 8; r++)
      C[(size_t)(row0 + half8 + r) * N + col] = acc[r] + bv;
  }
}

// ============================================================
// Pack an f32 matrix (K x N, arbitrary element strides) into bf16
// B-fragments: tile (ktile,ntile) -> 32 lanes x 16 bf16, each lane's
// 16 values contiguous (two b128 loads in the consumer).
// frag element i of lane L: k = ktile*32 + (L>>4)*8 + i + (i>=8 ? 8 : 0), n = ntile*16 + (L&15)
// Npad is a multiple of 64 for GEMM consumers (no ragged-edge branch).
// ============================================================
__global__ void pack_frag_kernel(const float* __restrict__ src, unsigned short* __restrict__ dst,
                                 int K, int N, int Npad, int strideK, int strideN,
                                 long srcStrideY, long dstStrideY,
                                 long srcStrideZ, long dstStrideZ) {
  const float* S = src + (size_t)blockIdx.y * srcStrideY + (size_t)blockIdx.z * srcStrideZ;
  unsigned short* D = dst + (size_t)blockIdx.y * dstStrideY + (size_t)blockIdx.z * dstStrideZ;
  const int nt16 = Npad >> 4;
  const int total = (K >> 5) * nt16 * 32;
  const int f = blockIdx.x * 256 + threadIdx.x;
  if (f >= total) return;
  const int L = f & 31, tile = f >> 5;
  const int ntile = tile % nt16, ktile = tile / nt16;
  const int n = ntile * 16 + (L & 15);
  const int kb = ktile * 32 + (L >> 4) * 8;
  union { unsigned short s[16]; uint4 q[2]; } tmp;
#pragma unroll
  for (int i = 0; i < 16; i++) {
    const float v = (n < N) ? S[(size_t)(kb + i + ((i >= 8) ? 8 : 0)) * strideK + (size_t)n * strideN] : 0.f;
    tmp.s[i] = f2bf(v);
  }
  uint4* d4 = (uint4*)(D + (size_t)f * 16);
  d4[0] = tmp.q[0];
  d4[1] = tmp.q[1];
}

// ============================================================
// WMMA GEMM: C[M,N] = A[M,K] @ W[K,N] (+bias), W pre-packed bf16 frags
// padded to Npad (multiple of 64) -> branch-free inner loop.
// ============================================================
__global__ void __launch_bounds__(32)
wmma_gemm_nn(const float* __restrict__ A, const unsigned short* __restrict__ Wp,
             const float* __restrict__ bias, float* __restrict__ C,
             int M, int N, int Npad, int K) {
  const int lane = threadIdx.x & 31;
  const int half = lane >> 4, mn = lane & 15, half8 = half * 8;
  const int row0 = blockIdx.y * 16;
  const int col0 = blockIdx.x * 64;
  const int nt16 = Npad >> 4;
  v8f acc0 = {}, acc1 = {}, acc2 = {}, acc3 = {};
  const float* arow = A + (size_t)(row0 + mn) * K + half8;
  const unsigned short* bbase = Wp + ((size_t)(col0 >> 4) * 32 + lane) * 16;
  const size_t bstep = (size_t)nt16 * 32 * 16;
  for (int k0 = 0; k0 < K; k0 += 32) {
    BFrag a;
    const float* ar = arow + k0;
    __builtin_prefetch(ar + 32, 0, 1);
#pragma unroll
    for (int i = 0; i < 8; i++) a.s[i]     = f2bf(ar[i]);
#pragma unroll
    for (int i = 0; i < 8; i++) a.s[8 + i] = f2bf(ar[16 + i]);
    BFrag b0, b1, b2, b3;
    load_bfrag(b0, bbase);
    load_bfrag(b1, bbase + 512);
    load_bfrag(b2, bbase + 1024);
    load_bfrag(b3, bbase + 1536);
    acc0 = wmma_bf16(a, b0, acc0);
    acc1 = wmma_bf16(a, b1, acc1);
    acc2 = wmma_bf16(a, b2, acc2);
    acc3 = wmma_bf16(a, b3, acc3);
    bbase += bstep;
  }
  store_tile(C, bias, acc0, row0, half8, col0 + mn,      N);
  store_tile(C, bias, acc1, row0, half8, col0 + 16 + mn, N);
  store_tile(C, bias, acc2, row0, half8, col0 + 32 + mn, N);
  store_tile(C, bias, acc3, row0, half8, col0 + 48 + mn, N);
}

// ============================================================
// CB[b,l,s] = sum_d Cm[b,l,d] * Bm[b,s,d]   (K = DS = 128)
// A = Cm rows (contiguous chunks), B = pre-packed Bm^T fragments.
// ============================================================
__global__ void __launch_bounds__(32)
cb_kernel(const float* __restrict__ xBCc, const unsigned short* __restrict__ bp,
          float* __restrict__ CB) {
  const int lane = threadIdx.x & 31;
  const int half = lane >> 4, mn = lane & 15, half8 = half * 8;
  const int s0 = blockIdx.x * 64, l0 = blockIdx.y * 16, b = blockIdx.z;
  const float* Cm = xBCc + (size_t)b * NTOK * CONVD + (DI_ + DS_);
  const unsigned short* bbase = bp + (size_t)b * (4 * 16 * 512)
                              + ((size_t)(s0 >> 4) * 32 + lane) * 16;
  v8f acc0 = {}, acc1 = {}, acc2 = {}, acc3 = {};
#pragma unroll
  for (int k0 = 0; k0 < DS_; k0 += 32) {
    BFrag a;
    const float* ar = Cm + (size_t)(l0 + mn) * CONVD + k0 + half8;
#pragma unroll
    for (int i = 0; i < 8; i++) a.s[i]     = f2bf(ar[i]);
#pragma unroll
    for (int i = 0; i < 8; i++) a.s[8 + i] = f2bf(ar[16 + i]);
    BFrag b0, b1, b2, b3;
    load_bfrag(b0, bbase);
    load_bfrag(b1, bbase + 512);
    load_bfrag(b2, bbase + 1024);
    load_bfrag(b3, bbase + 1536);
    acc0 = wmma_bf16(a, b0, acc0);
    acc1 = wmma_bf16(a, b1, acc1);
    acc2 = wmma_bf16(a, b2, acc2);
    acc3 = wmma_bf16(a, b3, acc3);
    bbase += (size_t)16 * 32 * 16;   // next k-tile (nt16 = 16)
  }
  float* out = CB + ((size_t)b * NTOK + l0 + half8) * NTOK + s0;
#pragma unroll
  for (int r = 0; r < 8; r++) {
    out[(size_t)r * NTOK + mn]      = acc0[r];
    out[(size_t)r * NTOK + 16 + mn] = acc1[r];
    out[(size_t)r * NTOK + 32 + mn] = acc2[r];
    out[(size_t)r * NTOK + 48 + mn] = acc3[r];
  }
}

// ============================================================
// Fused SSM attention: scores built on the fly (VALU exp) -> WMMA
// against pre-packed xs value fragments. One wave per (l-tile, h, b).
// ============================================================
__global__ void __launch_bounds__(32)
ssm_attn_kernel(const float* __restrict__ CB, const float* __restrict__ acum,
                const float* __restrict__ dt, const unsigned short* __restrict__ xsp,
                float* __restrict__ yv) {
  const int lane = threadIdx.x & 31;
  const int half = lane >> 4, mn = lane & 15, half8 = half * 8;
  const int l0 = blockIdx.x * 16, h = blockIdx.y, b = blockIdx.z;
  const int l  = l0 + mn;
  const float aL = acum[((size_t)b * NTOK + l) * NH_ + h];
  const float* cbrow = CB + ((size_t)b * NTOK + l) * NTOK;
  const unsigned short* xb = xsp + ((size_t)(b * NH_ + h)) * 16384 + (size_t)lane * 16;
  v8f acc0 = {}, acc1 = {}, acc2 = {}, acc3 = {};
  for (int s0 = 0; s0 < l0 + 16; s0 += 32) {
    BFrag a;
#pragma unroll
    for (int i = 0; i < 16; i++) {
      const int s = s0 + ((i >= 8) ? 16 : 0) + half8 + (i & 7);
      float v = 0.f;
      if (s <= l) {
        const float aS = acum[((size_t)b * NTOK + s) * NH_ + h];
        v = cbrow[s] * __expf(aL - aS) * dt[((size_t)b * NTOK + s) * NH_ + h];
      }
      a.s[i] = f2bf(v);
    }
    const unsigned short* bb0 = xb + (size_t)(s0 >> 5) * (4 * 32 * 16);  // nt16 = 4
    BFrag b0, b1, b2, b3;
    load_bfrag(b0, bb0);
    load_bfrag(b1, bb0 + 512);
    load_bfrag(b2, bb0 + 1024);
    load_bfrag(b3, bb0 + 1536);
    acc0 = wmma_bf16(a, b0, acc0);
    acc1 = wmma_bf16(a, b1, acc1);
    acc2 = wmma_bf16(a, b2, acc2);
    acc3 = wmma_bf16(a, b3, acc3);
  }
  float* out = yv + ((size_t)b * NTOK + l0 + half8) * DI_ + h * HD_;
#pragma unroll
  for (int r = 0; r < 8; r++) {
    out[(size_t)r * DI_ + mn]      = acc0[r];
    out[(size_t)r * DI_ + 16 + mn] = acc1[r];
    out[(size_t)r * DI_ + 32 + mn] = acc2[r];
    out[(size_t)r * DI_ + 48 + mn] = acc3[r];
  }
}

// ============================================================
// Patch embed for x and y, + 2D sincos POS; sy = silu(ye)
// ============================================================
__global__ void patch_embed_kernel(const float* __restrict__ x, const float* __restrict__ xw,
                                   const float* __restrict__ xb, const float* __restrict__ y,
                                   const float* __restrict__ yw, const float* __restrict__ yb,
                                   float* __restrict__ xe, float* __restrict__ sy) {
  const int idx = blockIdx.x * 256 + threadIdx.x;
  if (idx >= NB * NTOK * DM) return;
  const int d = idx % DM;
  const int tok = idx / DM;
  const int b = tok >> 8, n = tok & 255, g1 = n >> 4, g2 = n & 15;
  float ax = xb[d], ay = yb[d];
#pragma unroll
  for (int c = 0; c < 3; c++)
#pragma unroll
    for (int p1 = 0; p1 < 2; p1++)
#pragma unroll
      for (int p2 = 0; p2 < 2; p2++) {
        const int j = c * 4 + p1 * 2 + p2;
        const size_t src = ((size_t)(b * 3 + c) * 32 + (g1 * 2 + p1)) * 32 + (g2 * 2 + p2);
        ax += x[src] * xw[(size_t)j * DM + d];
        ay += y[src] * yw[(size_t)j * DM + d];
      }
  float posidx; int dd = d;
  if (d < 384) { posidx = (float)g2; } else { posidx = (float)g1; dd = d - 384; }
  const float lg = 9.21034037f / 192.0f;
  float pos;
  if (dd < 192) pos = sinf(posidx * __expf(-(float)dd * lg));
  else          pos = cosf(posidx * __expf(-(float)(dd - 192) * lg));
  ax += pos; ay += pos;
  xe[idx] = ax;
  sy[idx] = silu_f(ay);
}

// t-frequency embedding: emb[b, 0:128]=cos, [128:256]=sin
__global__ void tembed_emb_kernel(const float* __restrict__ t, float* __restrict__ emb) {
  const int idx = blockIdx.x * 256 + threadIdx.x;
  if (idx >= NB * 256) return;
  const int b = idx >> 8, j = idx & 255;
  const int k = (j < 128) ? j : (j - 128);
  const float f = __expf(-(float)k * (9.21034037f / 128.0f));
  const float a = t[b] * f;
  emb[idx] = (j < 128) ? cosf(a) : sinf(a);
}

// tiny GEMM (M<=4 rows): one thread per output
__global__ void small_gemm_kernel(const float* __restrict__ A, const float* __restrict__ W,
                                  const float* __restrict__ bias, float* __restrict__ C,
                                  int M, int N, int K, int act) {
  const int idx = blockIdx.x * 256 + threadIdx.x;
  if (idx >= M * N) return;
  const int m = idx / N, n = idx % N;
  float acc = bias ? bias[n] : 0.f;
  const float* a = A + (size_t)m * K;
  for (int k = 0; k < K; k++) acc += a[k] * W[(size_t)k * N + n];
  if (act == 1) acc = silu_f(acc);
  C[idx] = acc;
}

// LayerNorm + adaLN modulation
__global__ void ln_mod_kernel(const float* __restrict__ x, const float* __restrict__ tmod,
                              const float* __restrict__ ymod, float* __restrict__ out,
                              int tstride, int ystride) {
  const int tok = blockIdx.x;
  const int b = tok >> 8;
  const float* xr = x + (size_t)tok * DM;
  float s = 0.f, s2 = 0.f;
  for (int d = threadIdx.x; d < DM; d += 256) { float v = xr[d]; s += v; s2 += v * v; }
  const int lane = threadIdx.x & 31, wid = threadIdx.x >> 5;
  for (int o = 16; o > 0; o >>= 1) { s += __shfl_down(s, o); s2 += __shfl_down(s2, o); }
  __shared__ float sm[8], sm2[8];
  if (lane == 0) { sm[wid] = s; sm2[wid] = s2; }
  __syncthreads();
  if (threadIdx.x == 0) {
    float a = 0.f, c = 0.f;
    for (int i = 0; i < 8; i++) { a += sm[i]; c += sm2[i]; }
    sm[0] = a; sm2[0] = c;
  }
  __syncthreads();
  const float mean = sm[0] / (float)DM;
  const float var  = sm2[0] / (float)DM - mean * mean;
  const float rstd = rsqrtf(var + 1e-6f);
  const float* tr = tmod + (size_t)b * tstride;
  const float* yr = ymod + (size_t)tok * ystride;
  for (int d = threadIdx.x; d < DM; d += 256) {
    const float sh = tr[d] + yr[d];
    const float sc = tr[DM + d] + yr[DM + d];
    out[(size_t)tok * DM + d] = (xr[d] - mean) * rstd * (1.f + sc) + sh;
  }
}

// causal depthwise conv (k=4) + bias + silu
__global__ void dwconv_kernel(const float* __restrict__ zx, const float* __restrict__ w,
                              const float* __restrict__ bias, float* __restrict__ xBCc) {
  const int idx = blockIdx.x * 256 + threadIdx.x;
  if (idx >= NB * NTOK * CONVD) return;
  const int c = idx % CONVD;
  const int tok = idx / CONVD;
  const int n = tok & 255;
  float acc = bias[c];
#pragma unroll
  for (int k = 0; k < 4; k++) {
    const int nn = n - 3 + k;
    if (nn >= 0) acc += w[c * 4 + k] * zx[(size_t)(tok - n + nn) * DPROJ_ + DI_ + c];
  }
  xBCc[idx] = silu_f(acc);
}

// dt = softplus(dt_raw + bias); acum = cumsum(dt * -exp(A_log))
__global__ void dt_scan_kernel(const float* __restrict__ zx, const float* __restrict__ dt_bias,
                               const float* __restrict__ A_log, float* __restrict__ dt,
                               float* __restrict__ acum) {
  const int id = blockIdx.x * blockDim.x + threadIdx.x;
  if (id >= NB * NH_) return;
  const int b = id / NH_, h = id % NH_;
  const float negA = -__expf(A_log[h]);
  float acc = 0.f;
  for (int n = 0; n < NTOK; n++) {
    const size_t tok = (size_t)b * NTOK + n;
    const float raw = zx[tok * DPROJ_ + (DI_ + CONVD) + h] + dt_bias[h];
    const float d = (raw > 20.f) ? raw : log1pf(__expf(raw));
    dt[tok * NH_ + h] = d;
    acc += d * negA;
    acum[tok * NH_ + h] = acc;
  }
}

// yv = (y_ssm + D_ssm[h]*xs) * silu(z)
__global__ void ygate_kernel(float* __restrict__ yv, const float* __restrict__ xBCc,
                             const float* __restrict__ zx, const float* __restrict__ Dssm) {
  const int idx = blockIdx.x * 256 + threadIdx.x;
  if (idx >= NB * NTOK * DI_) return;
  const int i = idx % DI_;
  const int tok = idx / DI_;
  const int h = i >> 6;
  const float xs = xBCc[(size_t)tok * CONVD + i];
  const float z  = zx[(size_t)tok * DPROJ_ + i];
  yv[idx] = (yv[idx] + Dssm[h] * xs) * silu_f(z);
}

// xc += gate * mamba_out
__global__ void residual_kernel(float* __restrict__ xc, const float* __restrict__ mout,
                                const float* __restrict__ tmod, const float* __restrict__ ymod) {
  const int idx = blockIdx.x * 256 + threadIdx.x;
  if (idx >= NB * NTOK * DM) return;
  const int d = idx % DM;
  const int tok = idx / DM;
  const int b = tok >> 8;
  const float gate = tmod[(size_t)b * 3 * DM + 2 * DM + d] + ymod[(size_t)tok * 3 * DM + 2 * DM + d];
  xc[idx] += gate * mout[idx];
}

// unpatchify xf (B,N,36) -> img (B,9,32,32)
__global__ void unpatch_kernel(const float* __restrict__ xf, float* __restrict__ out) {
  const int idx = blockIdx.x * 256 + threadIdx.x;
  if (idx >= NB * OUTC_ * 32 * 32) return;
  const int ww = idx & 31, hh = (idx >> 5) & 31;
  const int c = (idx >> 10) % OUTC_;
  const int b = idx / (OUTC_ * 1024);
  const int g1 = hh >> 1, p1 = hh & 1, g2 = ww >> 1, p2 = ww & 1;
  const int n = g1 * 16 + g2;
  out[idx] = xf[(size_t)(b * NTOK + n) * (4 * OUTC_) + (p1 * 2 + p2) * OUTC_ + c];
}

// direct 3x3 SAME conv, optional relu / residual-add
__global__ void conv3_kernel(const float* __restrict__ in, const float* __restrict__ w,
                             const float* __restrict__ bias, float* __restrict__ out,
                             int Cin, int Cout, int relu, const float* __restrict__ resid) {
  const int idx = blockIdx.x * 256 + threadIdx.x;
  if (idx >= NB * Cout * 1024) return;
  const int xw = idx & 31, yh = (idx >> 5) & 31;
  const int co = (idx >> 10) % Cout;
  const int b = idx / (Cout * 1024);
  float acc = bias[co];
  for (int ci = 0; ci < Cin; ci++)
#pragma unroll
    for (int ky = -1; ky <= 1; ky++)
#pragma unroll
      for (int kx = -1; kx <= 1; kx++) {
        const int iy = yh + ky, ix = xw + kx;
        if (iy >= 0 && iy < 32 && ix >= 0 && ix < 32)
          acc += w[((size_t)(co * Cin + ci) * 3 + (ky + 1)) * 3 + (kx + 1)] *
                 in[((size_t)(b * Cin + ci) * 32 + iy) * 32 + ix];
      }
  if (resid) acc += resid[((size_t)(b * 3 + co) * 32 + yh) * 32 + xw];
  if (relu) acc = fmaxf(acc, 0.f);
  out[idx] = acc;
}

// ============================================================
static inline int packGrid(int K, int Npad) { return ((K >> 5) * (Npad >> 4) * 32 + 255) / 256; }

extern "C" void kernel_launch(void* const* d_in, const int* in_sizes, int n_in,
                              void* d_out, int out_size, void* d_ws, size_t ws_size,
                              hipStream_t stream) {
  const float* x      = (const float*)d_in[0];
  const float* t      = (const float*)d_in[1];
  const float* y      = (const float*)d_in[2];
  const float* Hs[3]  = { (const float*)d_in[3], (const float*)d_in[4], (const float*)d_in[5] };
  const float* xe_w   = (const float*)d_in[6];
  const float* xe_b   = (const float*)d_in[7];
  const float* ye_w   = (const float*)d_in[8];
  const float* ye_b   = (const float*)d_in[9];
  const float* tw1    = (const float*)d_in[10];
  const float* tb1    = (const float*)d_in[11];
  const float* tw2    = (const float*)d_in[12];
  const float* tb2    = (const float*)d_in[13];
  const float* adaln_w = (const float*)d_in[14];
  const float* adaln_b = (const float*)d_in[15];
  const float* in_w    = (const float*)d_in[16];
  const float* conv_w  = (const float*)d_in[17];
  const float* conv_b  = (const float*)d_in[18];
  const float* dt_bias = (const float*)d_in[19];
  const float* A_log   = (const float*)d_in[20];
  const float* D_ssm   = (const float*)d_in[21];
  const float* out_w   = (const float*)d_in[22];
  const float* fl_t_w  = (const float*)d_in[23];
  const float* fl_t_b  = (const float*)d_in[24];
  const float* fl_y_w  = (const float*)d_in[25];
  const float* fl_y_b  = (const float*)d_in[26];
  const float* fl_lin_w = (const float*)d_in[27];
  const float* fl_lin_b = (const float*)d_in[28];

  float* ws = (float*)d_ws;
  size_t off = 0;
  auto alloc = [&](size_t n) { float* p = ws + off; off += n; return p; };
  const size_t MTOK = (size_t)NB * NTOK;           // 1024
  float* xc   = alloc(MTOK * DM);
  float* sy   = alloc(MTOK * DM);
  float* emb  = alloc(NB * 256);
  float* hbuf = alloc(NB * DM);
  float* st   = alloc(NB * DM);
  float* tmod = alloc((size_t)NL_ * NB * 3 * DM);
  float* ymod = alloc(MTOK * 3 * DM);
  float* xm   = alloc(MTOK * DM);
  float* zx   = alloc(MTOK * DPROJ_);
  float* xBCc = alloc(MTOK * CONVD);
  float* dtb  = alloc(MTOK * NH_);
  float* acum = alloc(MTOK * NH_);
  float* CB   = alloc((size_t)NB * NTOK * NTOK);
  float* yv   = alloc(MTOK * DI_);
  float* mout = alloc(MTOK * DM);
  float* ft   = alloc(NB * 2 * DM);
  float* fy   = alloc(MTOK * 2 * DM);
  float* xf   = alloc(MTOK * 4 * OUTC_);
  float* c1   = alloc((size_t)NB * 64 * 1024);
  float* c2   = alloc((size_t)NB * 64 * 1024);
  unsigned short* wpack  = (unsigned short*)alloc((size_t)DM * 3392 / 2 + 16);  // max packed weight
  unsigned short* cbpack = (unsigned short*)alloc((size_t)NB * 32768 / 2 + 16);
  unsigned short* xspack = (unsigned short*)alloc((size_t)NB * NH_ * 16384 / 2 + 16);

  // ---- embeddings ----
  patch_embed_kernel<<<(MTOK * DM + 255) / 256, 256, 0, stream>>>(x, xe_w, xe_b, y, ye_w, ye_b, xc, sy);
  tembed_emb_kernel<<<(NB * 256 + 255) / 256, 256, 0, stream>>>(t, emb);
  small_gemm_kernel<<<(NB * DM + 255) / 256, 256, 0, stream>>>(emb, tw1, tb1, hbuf, NB, DM, 256, 1);
  small_gemm_kernel<<<(NB * DM + 255) / 256, 256, 0, stream>>>(hbuf, tw2, tb2, st, NB, DM, DM, 1);
  for (int l = 0; l < NL_; l++)
    small_gemm_kernel<<<(NB * 3 * DM + 255) / 256, 256, 0, stream>>>(
        st, adaln_w + (size_t)l * DM * 3 * DM, adaln_b + (size_t)l * 3 * DM,
        tmod + (size_t)l * NB * 3 * DM, NB, 3 * DM, DM, 0);

  // ---- mamba layers ----
  for (int l = 0; l < NL_; l++) {
    const float* ab = adaln_b + (size_t)l * 3 * DM;
    const float* tm = tmod + (size_t)l * NB * 3 * DM;
    // adaLN y-modulation GEMM (N = 2304, already /64)
    pack_frag_kernel<<<packGrid(DM, 3 * DM), 256, 0, stream>>>(
        adaln_w + (size_t)l * DM * 3 * DM, wpack, DM, 3 * DM, 3 * DM, 3 * DM, 1, 0, 0, 0, 0);
    wmma_gemm_nn<<<dim3((3 * DM) / 64, MTOK / 16), 32, 0, stream>>>(
        sy, wpack, ab, ymod, (int)MTOK, 3 * DM, 3 * DM, DM);
    ln_mod_kernel<<<MTOK, 256, 0, stream>>>(xc, tm, ymod, xm, 3 * DM, 3 * DM);
    // input projection GEMM (N = 3352, pad 3392 = 53*64)
    pack_frag_kernel<<<packGrid(DM, 3392), 256, 0, stream>>>(
        in_w + (size_t)l * DM * DPROJ_, wpack, DM, DPROJ_, 3392, DPROJ_, 1, 0, 0, 0, 0);
    wmma_gemm_nn<<<dim3(3392 / 64, MTOK / 16), 32, 0, stream>>>(
        xm, wpack, nullptr, zx, (int)MTOK, DPROJ_, 3392, DM);
    dwconv_kernel<<<(MTOK * CONVD + 255) / 256, 256, 0, stream>>>(
        zx, conv_w + (size_t)l * CONVD * 4, conv_b + (size_t)l * CONVD, xBCc);
    dt_scan_kernel<<<1, 128, 0, stream>>>(zx, dt_bias + l * NH_, A_log + l * NH_, dtb, acum);
    // pack Bm^T (K=128, N=256) per batch  and  xs (K=256 over s, N=64 over p) per (b,h)
    pack_frag_kernel<<<dim3(packGrid(DS_, NTOK), 1, NB), 256, 0, stream>>>(
        xBCc + DI_, cbpack, DS_, NTOK, NTOK, 1, CONVD,
        0, 0, (long)NTOK * CONVD, 32768);
    pack_frag_kernel<<<dim3(packGrid(NTOK, HD_), NH_, NB), 256, 0, stream>>>(
        xBCc, xspack, NTOK, HD_, HD_, CONVD, 1,
        HD_, 16384, (long)NTOK * CONVD, (long)NH_ * 16384);
    cb_kernel<<<dim3(NTOK / 64, NTOK / 16, NB), 32, 0, stream>>>(xBCc, cbpack, CB);
    ssm_attn_kernel<<<dim3(NTOK / 16, NH_, NB), 32, 0, stream>>>(CB, acum, dtb, xspack, yv);
    ygate_kernel<<<(MTOK * DI_ + 255) / 256, 256, 0, stream>>>(yv, xBCc, zx, D_ssm + l * NH_);
    // output projection GEMM
    pack_frag_kernel<<<packGrid(DI_, DM), 256, 0, stream>>>(
        out_w + (size_t)l * DI_ * DM, wpack, DI_, DM, DM, DM, 1, 0, 0, 0, 0);
    wmma_gemm_nn<<<dim3(DM / 64, MTOK / 16), 32, 0, stream>>>(
        yv, wpack, nullptr, mout, (int)MTOK, DM, DM, DI_);
    residual_kernel<<<(MTOK * DM + 255) / 256, 256, 0, stream>>>(xc, mout, tm, ymod);
  }

  // ---- final layer ----
  small_gemm_kernel<<<(NB * 2 * DM + 255) / 256, 256, 0, stream>>>(st, fl_t_w, fl_t_b, ft, NB, 2 * DM, DM, 0);
  pack_frag_kernel<<<packGrid(DM, 2 * DM), 256, 0, stream>>>(
      fl_y_w, wpack, DM, 2 * DM, 2 * DM, 2 * DM, 1, 0, 0, 0, 0);
  wmma_gemm_nn<<<dim3((2 * DM) / 64, MTOK / 16), 32, 0, stream>>>(
      sy, wpack, fl_y_b, fy, (int)MTOK, 2 * DM, 2 * DM, DM);
  ln_mod_kernel<<<MTOK, 256, 0, stream>>>(xc, ft, fy, xm, 2 * DM, 2 * DM);
  pack_frag_kernel<<<packGrid(DM, 64), 256, 0, stream>>>(
      fl_lin_w, wpack, DM, 4 * OUTC_, 64, 4 * OUTC_, 1, 0, 0, 0, 0);
  wmma_gemm_nn<<<dim3(1, MTOK / 16), 32, 0, stream>>>(
      xm, wpack, fl_lin_b, xf, (int)MTOK, 4 * OUTC_, 64, DM);
  unpatch_kernel<<<(NB * OUTC_ * 1024 + 255) / 256, 256, 0, stream>>>(xf, (float*)d_out);

  // ---- hfnet heads ----
  for (int i = 0; i < 3; i++) {
    const float* w1 = (const float*)d_in[29 + 6 * i + 0];
    const float* b1 = (const float*)d_in[29 + 6 * i + 1];
    const float* w2 = (const float*)d_in[29 + 6 * i + 2];
    const float* b2 = (const float*)d_in[29 + 6 * i + 3];
    const float* w3 = (const float*)d_in[29 + 6 * i + 4];
    const float* b3 = (const float*)d_in[29 + 6 * i + 5];
    float* op = (float*)d_out + NB * OUTC_ * 1024 + (size_t)i * NB * 3 * 1024;
    conv3_kernel<<<(NB * 64 * 1024 + 255) / 256, 256, 0, stream>>>(Hs[i], w1, b1, c1, 3, 64, 1, nullptr);
    conv3_kernel<<<(NB * 64 * 1024 + 255) / 256, 256, 0, stream>>>(c1, w2, b2, c2, 64, 64, 1, nullptr);
    conv3_kernel<<<(NB * 3 * 1024 + 255) / 256, 256, 0, stream>>>(c2, w3, b3, op, 64, 3, 0, Hs[i]);
  }
  (void)in_sizes; (void)n_in; (void)out_size; (void)ws_size;
}
